// PagAFT_74826920231628
// MI455X (gfx1250) — matrix operand order repaired and checked by reading, not compiled, not run on hardware
//
#include <hip/hip_runtime.h>

// ---------------------------------------------------------------------------
// PagAFT on MI455X (gfx1250): bf16 WMMA GEMMs + Tensor Data Mover staging.
// Pipeline (channel-major so x/y/out need no transposes):
//   1) cvt      : x,y,Wq,Wk,Wv  f32 -> bf16
//   2) expT     : ebT[j][i] = bf16(exp(pos_bias[i][j]))  (tiled LDS transpose)
//   3) qkv_gemm : qT/kT/vT[b] = W @ x[b]/y[b] (+bias) -> sigq(f32), ek, v
//   4) mul_ekv  : ekv = ek * v (bf16)
//   5) aft_gemm : numT = ekvT@ebT, denT = ekT@ebT fused; out = sigq*num/den
// GEMM kernels: 64x128 workgroup tile, 8 waves x (32x32), K-step 32,
// double-buffered LDS; A tiles staged by TDM (tensor_load_to_lds, 6-arg
// clang-23 form), B tiles staged transposed with packed b32 DS stores.
// ---------------------------------------------------------------------------

typedef __bf16 bf16;
typedef __bf16 v8bf  __attribute__((ext_vector_type(8)));
typedef __bf16 v16bf __attribute__((ext_vector_type(16)));
typedef float  v8f   __attribute__((ext_vector_type(8)));
typedef unsigned short v8u16 __attribute__((ext_vector_type(8)));

constexpr int kB = 8;
constexpr int kC = 512;
constexpr int kN = 2304;   // H*W = 48*48

#if __has_builtin(__builtin_amdgcn_tensor_load_to_lds) && \
    __has_builtin(__builtin_amdgcn_s_wait_tensorcnt)
#define USE_TDM 1
#endif

#ifdef USE_TDM
typedef unsigned int u32x4 __attribute__((ext_vector_type(4)));
typedef int          i32x8 __attribute__((ext_vector_type(8)));
typedef int          i32x4 __attribute__((ext_vector_type(4)));

// Issue a TDM 2-D tile load (bf16 elements) global -> LDS.
// Tile rows are tile_w elements (64B here); LDS rows are padded by 16B
// (pad_interval=3 -> every 16 DWORDs, pad_amount=3 -> 4 DWORDs), matching
// the 40-element padded LDS row stride used by the compute code.
__device__ __forceinline__ void tdm_load_2d_bf16(unsigned lds_off,
                                                 const bf16* gptr,
                                                 unsigned tile_w, unsigned tile_h,
                                                 unsigned tensor_w, unsigned tensor_h,
                                                 unsigned row_stride_elems) {
  const unsigned long long ga = (unsigned long long)(uintptr_t)gptr;
  u32x4 g0;
  g0.x = 1u;                                  // count=1 (valid user descriptor)
  g0.y = lds_off;                             // LDS byte address
  g0.z = (unsigned)ga;                        // global_addr[31:0]
  g0.w = (unsigned)((ga >> 32) & 0x01FFFFFFu) | (2u << 30);  // addr[56:32]|type=2
  i32x8 g1;
  // data_size=1 (2B), pad_enable=1, pad_interval=3 (16 DW), pad_amount=3 (4 DW)
  g1[0] = (int)((1u << 16) | (1u << 20) | (3u << 22) | (3u << 25));
  g1[1] = (int)((tensor_w & 0xFFFFu) << 16);                    // dim0[15:0]
  g1[2] = (int)((tensor_w >> 16) | ((tensor_h & 0xFFFFu) << 16));
  g1[3] = (int)((tensor_h >> 16) | (tile_w << 16));             // tile_dim0
  g1[4] = (int)(tile_h & 0xFFFFu);                              // tile_dim1
  g1[5] = (int)row_stride_elems;                                // dim0_stride
  g1[6] = 0;
  g1[7] = 0;
  i32x4 g2 = {0, 0, 0, 0};
  i32x4 g3 = {0, 0, 0, 0};
  i32x8 g4 = {0, 0, 0, 0, 0, 0, 0, 0};
  __builtin_amdgcn_tensor_load_to_lds(g0, g1, g2, g3, g4, 0);
}
#endif

// Concatenate two 8-wide bf16 vectors into one 16-wide WMMA fragment.
__device__ __forceinline__ v16bf concat8(v8bf lo, v8bf hi) {
  return __builtin_shufflevector(lo, hi, 0, 1, 2, 3, 4, 5, 6, 7,
                                         8, 9, 10, 11, 12, 13, 14, 15);
}

// ---------------------------------------------------------------------------
// 1) f32 -> bf16 conversion
// ---------------------------------------------------------------------------
__global__ void cvt_f32_bf16(const float* __restrict__ in,
                             bf16* __restrict__ out, int n) {
  int i = blockIdx.x * blockDim.x + threadIdx.x;
  if (i < n) out[i] = (bf16)in[i];
}

// ---------------------------------------------------------------------------
// 2) ebT[j][i] = bf16(exp(pos_bias[i][j]))  -- 32x32 LDS-tiled transpose
// ---------------------------------------------------------------------------
__global__ __launch_bounds__(256)
void exp_transpose(const float* __restrict__ pos, bf16* __restrict__ ebT) {
  __shared__ float tile[32][33];
  const int i0 = blockIdx.y * 32, j0 = blockIdx.x * 32;
  const int tx = threadIdx.x, ty = threadIdx.y;
#pragma unroll
  for (int r = 0; r < 32; r += 8)
    tile[ty + r][tx] = pos[(size_t)(i0 + ty + r) * kN + (j0 + tx)];
  __syncthreads();
#pragma unroll
  for (int r = 0; r < 32; r += 8)
    ebT[(size_t)(j0 + ty + r) * kN + (i0 + tx)] = (bf16)__expf(tile[tx][ty + r]);
}

// ---------------------------------------------------------------------------
// 3) QKV projection GEMM (NN, channel-major):
//    outT[c_out, i] = sum_cin W[c_out, cin] * X[cin, i] + bias[c_out]
//    blockIdx.z = b*3 + t  (t: 0=q, 1=k, 2=v)
// ---------------------------------------------------------------------------
__global__ __launch_bounds__(256)
void qkv_gemm(const bf16* __restrict__ wqb, const bf16* __restrict__ wkb,
              const bf16* __restrict__ wvb,
              const float* __restrict__ bq, const float* __restrict__ bk,
              const float* __restrict__ bv,
              const bf16* __restrict__ xb, const bf16* __restrict__ yb,
              float* __restrict__ sigq, bf16* __restrict__ ekb,
              bf16* __restrict__ vbuf) {
  __shared__ bf16 As[2][64][40];    // [buf] A tile, K-contiguous rows
  __shared__ bf16 Bs[2][128][40];   // [buf] B tile transposed: Bs[.][n][k]

  const int t = blockIdx.z % 3;
  const int b = blockIdx.z / 3;
  const bf16* __restrict__ W     = (t == 0) ? wqb : (t == 1) ? wkb : wvb;
  const float* __restrict__ bias = (t == 0) ? bq : (t == 1) ? bk : bv;
  const bf16* __restrict__ X = ((t == 0) ? xb : yb) + (size_t)b * kC * kN;

  const int m0 = blockIdx.y * 64;
  const int n0 = blockIdx.x * 128;
  const int tid = threadIdx.x;
  const int wave = tid >> 5, lane = tid & 31;
  const int l16 = lane & 15, half = lane >> 4;
  const int waveM = wave >> 2, waveN = wave & 3;   // 2x4 wave grid
  const int p = tid >> 4, cb = tid & 15;           // B staging task

  // Stage B: load two consecutive K-rows, pack K-pairs into dwords,
  // store transposed with 8 x ds_store_b32.
  auto stageB = [&](int kt, int buf) {
    v8u16 r0 = __builtin_bit_cast(
        v8u16, *(const v8bf*)&X[(size_t)(kt + 2 * p) * kN + n0 + cb * 8]);
    v8u16 r1 = __builtin_bit_cast(
        v8u16, *(const v8bf*)&X[(size_t)(kt + 2 * p + 1) * kN + n0 + cb * 8]);
#pragma unroll
    for (int j = 0; j < 8; ++j)
      *(unsigned*)&Bs[buf][cb * 8 + j][2 * p] =
          (unsigned)r0[j] | ((unsigned)r1[j] << 16);
  };
  auto stageA = [&](int kt, int buf) {
#ifdef USE_TDM
    if (wave == 0)
      tdm_load_2d_bf16((unsigned)(uintptr_t)&As[buf][0][0],
                       &W[(size_t)m0 * kC + kt], 32, 64,
                       (unsigned)(kC - kt), (unsigned)(kC - m0), kC);
#else
    const int row = tid >> 2, q = tid & 3;
    *(v8bf*)&As[buf][row][q * 8] =
        *(const v8bf*)&W[(size_t)(m0 + row) * kC + kt + q * 8];
#endif
  };

  v8f acc[2][2] = {};
  stageB(0, 0);
  stageA(0, 0);
#ifdef USE_TDM
  if (wave == 0) __builtin_amdgcn_s_wait_tensorcnt(0);
#endif
  __syncthreads();

  int ibuf = 0;
  for (int kt = 0; kt < kC; kt += 32) {
    const int nkt = kt + 32;
    if (nkt < kC) {       // prefetch next tiles into the other buffer
      stageB(nkt, ibuf ^ 1);
      stageA(nkt, ibuf ^ 1);
    }
    v16bf af[2], bfr[2];
#pragma unroll
    for (int tm = 0; tm < 2; ++tm) {
      const bf16* pa = &As[ibuf][waveM * 32 + tm * 16 + l16][half * 8];
      af[tm] = concat8(*(const v8bf*)pa, *(const v8bf*)(pa + 16));
    }
#pragma unroll
    for (int tn = 0; tn < 2; ++tn) {
      const bf16* pb = &Bs[ibuf][waveN * 32 + tn * 16 + l16][half * 8];
      bfr[tn] = concat8(*(const v8bf*)pb, *(const v8bf*)(pb + 16));
    }
#pragma unroll
    for (int tm = 0; tm < 2; ++tm)
#pragma unroll
      for (int tn = 0; tn < 2; ++tn)
        acc[tm][tn] = __builtin_amdgcn_wmma_f32_16x16x32_bf16(
            false, af[tm], false, bfr[tn], (short)0, acc[tm][tn], false, false);
#ifdef USE_TDM
    if (wave == 0) __builtin_amdgcn_s_wait_tensorcnt(0);
#endif
    __syncthreads();
    ibuf ^= 1;
  }

  // Epilogue: add bias; per-projection nonlinearity + store (channel-major).
#pragma unroll
  for (int tm = 0; tm < 2; ++tm) {
#pragma unroll
    for (int tn = 0; tn < 2; ++tn) {
#pragma unroll
      for (int r = 0; r < 8; ++r) {
        const int m = m0 + waveM * 32 + tm * 16 + r + half * 8;
        const int nn = n0 + waveN * 32 + tn * 16 + l16;
        const float val = acc[tm][tn][r] + bias[m];
        const size_t idx = ((size_t)b * kC + m) * kN + nn;
        if (t == 0)      sigq[idx] = 1.0f / (1.0f + __expf(-val));
        else if (t == 1) ekb[idx]  = (bf16)__expf(val);
        else             vbuf[idx] = (bf16)val;
      }
    }
  }
}

// ---------------------------------------------------------------------------
// 4) ekv = ek * v (elementwise, bf16)
// ---------------------------------------------------------------------------
__global__ void mul_ekv(const bf16* __restrict__ ek, const bf16* __restrict__ v,
                        bf16* __restrict__ ekv, int n) {
  int i = blockIdx.x * blockDim.x + threadIdx.x;
  if (i < n) ekv[i] = (bf16)((float)ek[i] * (float)v[i]);
}

// ---------------------------------------------------------------------------
// 5) Fused AFT GEMM: numT = ekvT@ebT, denT = ekT@ebT (shared B tile),
//    out[b,c,i] = sigq[b,c,i] * numT/denT  directly in (B,C,H,W).
// ---------------------------------------------------------------------------
__global__ __launch_bounds__(256)
void aft_gemm(const bf16* __restrict__ ekvb, const bf16* __restrict__ ekb,
              const bf16* __restrict__ ebT, const float* __restrict__ sigq,
              float* __restrict__ out) {
  __shared__ bf16 As[2][2][64][40];  // [buf][0]=ekvT tile, [buf][1]=ekT tile
  __shared__ bf16 Bs[2][128][40];    // [buf] ebT tile transposed

  const int b = blockIdx.z;
  const bf16* __restrict__ A0 = ekvb + (size_t)b * kC * kN;
  const bf16* __restrict__ A1 = ekb  + (size_t)b * kC * kN;
  const int m0 = blockIdx.y * 64;
  const int n0 = blockIdx.x * 128;
  const int tid = threadIdx.x;
  const int wave = tid >> 5, lane = tid & 31;
  const int l16 = lane & 15, half = lane >> 4;
  const int waveM = wave >> 2, waveN = wave & 3;
  const int p = tid >> 4, cb = tid & 15;

  auto stageB = [&](int kt, int buf) {
    v8u16 r0 = __builtin_bit_cast(
        v8u16, *(const v8bf*)&ebT[(size_t)(kt + 2 * p) * kN + n0 + cb * 8]);
    v8u16 r1 = __builtin_bit_cast(
        v8u16, *(const v8bf*)&ebT[(size_t)(kt + 2 * p + 1) * kN + n0 + cb * 8]);
#pragma unroll
    for (int j = 0; j < 8; ++j)
      *(unsigned*)&Bs[buf][cb * 8 + j][2 * p] =
          (unsigned)r0[j] | ((unsigned)r1[j] << 16);
  };
  auto stageA = [&](int kt, int buf) {
#ifdef USE_TDM
    if (wave == 0) {
      tdm_load_2d_bf16((unsigned)(uintptr_t)&As[buf][0][0][0],
                       &A0[(size_t)m0 * kN + kt], 32, 64,
                       (unsigned)(kN - kt), (unsigned)(kC - m0), kN);
      tdm_load_2d_bf16((unsigned)(uintptr_t)&As[buf][1][0][0],
                       &A1[(size_t)m0 * kN + kt], 32, 64,
                       (unsigned)(kN - kt), (unsigned)(kC - m0), kN);
    }
#else
    const int row = tid >> 2, q = tid & 3;
    const size_t off = (size_t)(m0 + row) * kN + kt + q * 8;
    *(v8bf*)&As[buf][0][row][q * 8] = *(const v8bf*)&A0[off];
    *(v8bf*)&As[buf][1][row][q * 8] = *(const v8bf*)&A1[off];
#endif
  };

  v8f accN[2][2] = {};
  v8f accD[2][2] = {};
  stageB(0, 0);
  stageA(0, 0);
#ifdef USE_TDM
  if (wave == 0) __builtin_amdgcn_s_wait_tensorcnt(0);
#endif
  __syncthreads();

  int ibuf = 0;
  for (int kt = 0; kt < kN; kt += 32) {
    const int nkt = kt + 32;
    if (nkt < kN) {
      stageB(nkt, ibuf ^ 1);
      stageA(nkt, ibuf ^ 1);
    }
    v16bf a0[2], a1[2], bfr[2];
#pragma unroll
    for (int tm = 0; tm < 2; ++tm) {
      const bf16* p0 = &As[ibuf][0][waveM * 32 + tm * 16 + l16][half * 8];
      const bf16* p1 = &As[ibuf][1][waveM * 32 + tm * 16 + l16][half * 8];
      a0[tm] = concat8(*(const v8bf*)p0, *(const v8bf*)(p0 + 16));
      a1[tm] = concat8(*(const v8bf*)p1, *(const v8bf*)(p1 + 16));
    }
#pragma unroll
    for (int tn = 0; tn < 2; ++tn) {
      const bf16* pb = &Bs[ibuf][waveN * 32 + tn * 16 + l16][half * 8];
      bfr[tn] = concat8(*(const v8bf*)pb, *(const v8bf*)(pb + 16));
    }
#pragma unroll
    for (int tm = 0; tm < 2; ++tm)
#pragma unroll
      for (int tn = 0; tn < 2; ++tn) {
        accN[tm][tn] = __builtin_amdgcn_wmma_f32_16x16x32_bf16(
            false, a0[tm], false, bfr[tn], (short)0, accN[tm][tn], false, false);
        accD[tm][tn] = __builtin_amdgcn_wmma_f32_16x16x32_bf16(
            false, a1[tm], false, bfr[tn], (short)0, accD[tm][tn], false, false);
      }
#ifdef USE_TDM
    if (wave == 0) __builtin_amdgcn_s_wait_tensorcnt(0);
#endif
    __syncthreads();
    ibuf ^= 1;
  }

#pragma unroll
  for (int tm = 0; tm < 2; ++tm) {
#pragma unroll
    for (int tn = 0; tn < 2; ++tn) {
#pragma unroll
      for (int r = 0; r < 8; ++r) {
        const int m = m0 + waveM * 32 + tm * 16 + r + half * 8;
        const int nn = n0 + waveN * 32 + tn * 16 + l16;
        const size_t idx = ((size_t)b * kC + m) * kN + nn;
        out[idx] = sigq[idx] * (accN[tm][tn][r] / accD[tm][tn][r]);
      }
    }
  }
}

// ---------------------------------------------------------------------------
// Host launcher
// ---------------------------------------------------------------------------
extern "C" void kernel_launch(void* const* d_in, const int* in_sizes, int n_in,
                              void* d_out, int out_size, void* d_ws,
                              size_t ws_size, hipStream_t stream) {
  (void)in_sizes; (void)n_in; (void)out_size; (void)ws_size;
  const float* x   = (const float*)d_in[0];
  const float* y   = (const float*)d_in[1];
  const float* Wq  = (const float*)d_in[2];
  const float* bq  = (const float*)d_in[3];
  const float* Wk  = (const float*)d_in[4];
  const float* bk  = (const float*)d_in[5];
  const float* Wv  = (const float*)d_in[6];
  const float* bv  = (const float*)d_in[7];
  const float* pos = (const float*)d_in[8];
  float* out = (float*)d_out;

  const size_t nBCN = (size_t)kB * kC * kN;   // 9,437,184
  const size_t nCC  = (size_t)kC * kC;        // 262,144
  const size_t nNN  = (size_t)kN * kN;        // 5,308,416

  char* w = (char*)d_ws;
  auto carve = [&](size_t bytes) {
    char* pc = w;
    w += (bytes + 255) & ~(size_t)255;
    return pc;
  };
  bf16* xb    = (bf16*)carve(nBCN * 2);
  bf16* yb    = (bf16*)carve(nBCN * 2);
  bf16* wqb   = (bf16*)carve(nCC * 2);
  bf16* wkb   = (bf16*)carve(nCC * 2);
  bf16* wvb   = (bf16*)carve(nCC * 2);
  bf16* ebT   = (bf16*)carve(nNN * 2);
  float* sigq = (float*)carve(nBCN * 4);
  bf16* ekb   = (bf16*)carve(nBCN * 2);
  bf16* vbuf  = (bf16*)carve(nBCN * 2);
  bf16* ekvb  = (bf16*)carve(nBCN * 2);

  cvt_f32_bf16<<<(int)((nBCN + 255) / 256), 256, 0, stream>>>(x, xb, (int)nBCN);
  cvt_f32_bf16<<<(int)((nBCN + 255) / 256), 256, 0, stream>>>(y, yb, (int)nBCN);
  cvt_f32_bf16<<<(int)((nCC + 255) / 256), 256, 0, stream>>>(Wq, wqb, (int)nCC);
  cvt_f32_bf16<<<(int)((nCC + 255) / 256), 256, 0, stream>>>(Wk, wkb, (int)nCC);
  cvt_f32_bf16<<<(int)((nCC + 255) / 256), 256, 0, stream>>>(Wv, wvb, (int)nCC);

  exp_transpose<<<dim3(kN / 32, kN / 32), dim3(32, 8), 0, stream>>>(pos, ebT);

  qkv_gemm<<<dim3(kN / 128, kC / 64, 3 * kB), 256, 0, stream>>>(
      wqb, wkb, wvb, bq, bk, bv, xb, yb, sigq, ekb, vbuf);

  mul_ekv<<<(int)((nBCN + 255) / 256), 256, 0, stream>>>(ekb, vbuf, ekvb,
                                                         (int)nBCN);

  aft_gemm<<<dim3(kN / 128, kC / 64, kB), 256, 0, stream>>>(ekvb, ekb, ebT,
                                                            sigq, out);
}
